// SmolLM2Attention_29635274342952
// MI455X (gfx1250) — matrix operand order repaired and checked by reading, not compiled, not run on hardware
//
#include <hip/hip_runtime.h>
#include <hip/hip_bf16.h>

#define HIDDEN   2048
#define N_HEADS  32
#define N_KV     8
#define HEAD_DIM 64
#define S_LEN    2048
#define BATCH    2
#define QSIZE    (N_HEADS * HEAD_DIM)      // 2048
#define KVSIZE   (N_KV * HEAD_DIM)         // 512
#define E_QKV    (QSIZE + 2 * KVSIZE)      // 3072
#define NTOK     (BATCH * S_LEN)           // 4096

typedef __bf16 bf16_t;
typedef bf16_t       bf16x16 __attribute__((ext_vector_type(16)));
typedef float        f32x8   __attribute__((ext_vector_type(8)));
typedef unsigned int u32x4   __attribute__((ext_vector_type(4)));

union FragU {
    bf16x16 v;
    u32x4   q[2];
    unsigned short u[16];
};

__device__ inline unsigned short f2bf(float f) {
    unsigned int u = __float_as_uint(f);
    u += 0x7FFFu + ((u >> 16) & 1u);   // round-to-nearest-even
    return (unsigned short)(u >> 16);
}

// A fragment (16x32 bf16, MxK): lane row m0+l%16; e0..7 -> K=k0+hf*8+e, e8..15 -> K=k0+16+hf*8+(e-8)
__device__ inline bf16x16 load_a_frag_g(const unsigned short* A, int lda, int m0, int k0) {
    int lane = threadIdx.x & 31;
    const unsigned short* p = A + (size_t)(m0 + (lane & 15)) * lda + k0 + ((lane >> 4) << 3);
    FragU f;
    f.q[0] = *(const u32x4*)(p);        // K = k0 + hf*8 .. +7
    f.q[1] = *(const u32x4*)(p + 16);   // K = k0 + 16 + hf*8 .. +7
    return f.v;
}

// B fragment (32x16 bf16, KxN) from row-major B^T (rows = N, cols = K):
// lane row n0+l%16; e0..7 -> K=k0+hf*8+e, e8..15 -> K=k0+16+hf*8+(e-8)
__device__ inline bf16x16 load_b_frag_g(const unsigned short* Bt, int ldb, int n0, int k0) {
    int lane = threadIdx.x & 31;
    const unsigned short* p = Bt + (size_t)(n0 + (lane & 15)) * ldb + k0 + ((lane >> 4) << 3);
    FragU f;
    f.q[0] = *(const u32x4*)(p);
    f.q[1] = *(const u32x4*)(p + 16);
    return f.v;
}

__device__ inline f32x8 wmma_bf16(bf16x16 a, bf16x16 b, f32x8 c) {
    return __builtin_amdgcn_wmma_f32_16x16x32_bf16(false, a, false, b, (short)0, c, false, false);
}

// ---------------- f32 -> bf16 conversion ----------------
__global__ void __launch_bounds__(256, 1)
k_f32_to_bf16(const float* __restrict__ in, unsigned short* __restrict__ out, int n) {
    int i = blockIdx.x * blockDim.x + threadIdx.x;
    int stride = gridDim.x * blockDim.x;
    for (; i < n; i += stride) out[i] = f2bf(in[i]);
}

// ---------------- GEMM: C[M,N] = A[M,K](bf16) * W[N,K](bf16)^T, f32 out ----------------
// One wave computes a 64x64 tile: 16 accumulators, 16 WMMAs per 32-wide K step.
// B-frags are loaded one at a time to keep live VGPRs ~180 (no spills with launch_bounds(128,1)).
__global__ void __launch_bounds__(128, 1)
k_gemm_bf16(const unsigned short* __restrict__ A,
            const unsigned short* __restrict__ W,
            float* __restrict__ C, int M, int N, int K) {
    int wid = blockIdx.x * (blockDim.x >> 5) + (threadIdx.x >> 5);
    int tn_count = N >> 6;
    int tm = wid / tn_count, tn = wid - tm * tn_count;
    int m0 = tm << 6, n0 = tn << 6;
    if (m0 >= M) return;

    f32x8 acc[4][4] = {};
    for (int k0 = 0; k0 < K; k0 += 32) {
        bf16x16 a[4];
#pragma unroll
        for (int i = 0; i < 4; i++) a[i] = load_a_frag_g(A, K, m0 + i * 16, k0);
#pragma unroll
        for (int j = 0; j < 4; j++) {
            bf16x16 b = load_b_frag_g(W, K, n0 + j * 16, k0);
#pragma unroll
            for (int i = 0; i < 4; i++)
                acc[i][j] = wmma_bf16(a[i], b, acc[i][j]);
        }
    }
    int lane = threadIdx.x & 31, hf = lane >> 4, nn = lane & 15;
#pragma unroll
    for (int i = 0; i < 4; i++)
#pragma unroll
        for (int j = 0; j < 4; j++)
#pragma unroll
            for (int r = 0; r < 8; r++)
                C[(size_t)(m0 + i * 16 + r + hf * 8) * N + (n0 + j * 16 + nn)] = acc[i][j][r];
}

// ---------------- per-head RMSNorm + neox RoPE + layout shuffle ----------------
// One wave per (token, unit): units 0..31 = Q heads, 32..39 = K heads, 40..47 = V heads.
__global__ void __launch_bounds__(256, 1)
k_norm_rope(const int* __restrict__ positions,
            const float* __restrict__ qkv,
            const float* __restrict__ qw,
            const float* __restrict__ kw,
            unsigned short* __restrict__ Qb,   // [B,H,S,64]
            unsigned short* __restrict__ Kb,   // [B,KV,S,64]
            unsigned short* __restrict__ Vt)   // [B,KV,64,S] (transposed)
{
    int wid = blockIdx.x * (blockDim.x >> 5) + (threadIdx.x >> 5);
    int t = wid / 48, j = wid - t * 48;
    if (t >= NTOK) return;
    int b = t >> 11, s = t & (S_LEN - 1);
    int lane = threadIdx.x & 31;

    const float* row = qkv + (size_t)t * E_QKV;
    int off = (j < 32) ? j * 64 : (j < 40 ? QSIZE + (j - 32) * 64 : QSIZE + KVSIZE + (j - 40) * 64);
    float x1 = row[off + lane];
    float x2 = row[off + 32 + lane];

    if (j < 40) {
        float ss = x1 * x1 + x2 * x2;
#pragma unroll
        for (int m = 16; m >= 1; m >>= 1) ss += __shfl_xor(ss, m, 32);
        float inv = rsqrtf(ss * (1.0f / 64.0f) + 1e-6f);
        const float* w = (j < 32) ? qw : kw;
        x1 *= inv * w[lane];
        x2 *= inv * w[32 + lane];
        // inv_freq = 10000^(-lane/32); ln(10000)/32 = 0.28782313662425572
        float pos = (float)positions[s];
        float fr = pos * __expf(-0.28782313662425572f * (float)lane);
        float sn, cs;
        sincosf(fr, &sn, &cs);   // accurate range reduction (args up to ~2047 rad)
        float y1 = x1 * cs - x2 * sn;
        float y2 = x2 * cs + x1 * sn;
        if (j < 32) {
            y1 *= 0.125f; y2 *= 0.125f;   // fold head_dim^-0.5 into Q
            size_t base = ((size_t)(b * N_HEADS + j) * S_LEN + s) * HEAD_DIM;
            Qb[base + lane]      = f2bf(y1);
            Qb[base + 32 + lane] = f2bf(y2);
        } else {
            size_t base = ((size_t)(b * N_KV + (j - 32)) * S_LEN + s) * HEAD_DIM;
            Kb[base + lane]      = f2bf(y1);
            Kb[base + 32 + lane] = f2bf(y2);
        }
    } else {
        int kv = j - 40;
        size_t base = (size_t)(b * N_KV + kv) * HEAD_DIM * S_LEN;
        Vt[base + (size_t)lane * S_LEN + s]        = f2bf(x1);
        Vt[base + (size_t)(lane + 32) * S_LEN + s] = f2bf(x2);
    }
}

// ---------------- causal flash attention, one wave per (b, h, 16 q-rows) ----------------
#define ATT_WAVES 4
__global__ void __launch_bounds__(128, 1)
k_attention(const unsigned short* __restrict__ Qb,
            const unsigned short* __restrict__ Kb,
            const unsigned short* __restrict__ Vt,
            unsigned short* __restrict__ Ob)   // [NTOK, QSIZE]
{
    __shared__ __align__(16) unsigned short plds[ATT_WAVES][16 * 32];
    int warp = threadIdx.x >> 5;
    int wid = blockIdx.x * ATT_WAVES + warp;
    int qt = wid & 127;          // S/16
    int bh = wid >> 7;
    int h = bh & (N_HEADS - 1);
    int b = bh >> 5;
    int kvh = h >> 2;            // GQA group of 4

    const unsigned short* Qh = Qb + (size_t)(b * N_HEADS + h) * S_LEN * HEAD_DIM;
    const unsigned short* Kh = Kb + (size_t)(b * N_KV + kvh) * S_LEN * HEAD_DIM;
    const unsigned short* Vh = Vt + (size_t)(b * N_KV + kvh) * HEAD_DIM * S_LEN;
    int lane = threadIdx.x & 31, hf = lane >> 4, nn = lane & 15;
    int qbase = qt * 16;
    unsigned short* lp = plds[warp];

    bf16x16 qa0 = load_a_frag_g(Qh, HEAD_DIM, qbase, 0);
    bf16x16 qa1 = load_a_frag_g(Qh, HEAD_DIM, qbase, 32);

    f32x8 o[4] = {};
    float mrow[8], lrow[8];
#pragma unroll
    for (int r = 0; r < 8; r++) { mrow[r] = -3.0e38f; lrow[r] = 0.0f; }

    int nchunk = (qbase + 16 + 31) >> 5;
    for (int kb = 0; kb < nchunk; ++kb) {
        int kbase = kb << 5;
        f32x8 s0 = {}, s1 = {};
        s0 = wmma_bf16(qa0, load_b_frag_g(Kh, HEAD_DIM, kbase,      0),  s0);
        s0 = wmma_bf16(qa1, load_b_frag_g(Kh, HEAD_DIM, kbase,      32), s0);
        s1 = wmma_bf16(qa0, load_b_frag_g(Kh, HEAD_DIM, kbase + 16, 0),  s1);
        s1 = wmma_bf16(qa1, load_b_frag_g(Kh, HEAD_DIM, kbase + 16, 32), s1);

        if (kbase + 31 > qbase) {   // chunk touches the diagonal
#pragma unroll
            for (int r = 0; r < 8; r++) {
                int qrow = qbase + r + hf * 8;
                if (kbase + nn > qrow)      s0[r] = -3.0e38f;
                if (kbase + 16 + nn > qrow) s1[r] = -3.0e38f;
            }
        }
#pragma unroll
        for (int r = 0; r < 8; r++) {
            float mx = fmaxf(s0[r], s1[r]);
#pragma unroll
            for (int m = 8; m >= 1; m >>= 1) mx = fmaxf(mx, __shfl_xor(mx, m, 32));
            float mnew = fmaxf(mrow[r], mx);
            float scale = __expf(mrow[r] - mnew);
            float p0 = __expf(s0[r] - mnew);
            float p1 = __expf(s1[r] - mnew);
            float rs = p0 + p1;
#pragma unroll
            for (int m = 8; m >= 1; m >>= 1) rs += __shfl_xor(rs, m, 32);
            lrow[r] = lrow[r] * scale + rs;
            mrow[r] = mnew;
#pragma unroll
            for (int t = 0; t < 4; t++) o[t][r] *= scale;
            int lr = r + hf * 8;
            lp[lr * 32 + nn]      = f2bf(p0);
            lp[lr * 32 + 16 + nn] = f2bf(p1);
        }
        asm volatile("s_wait_dscnt 0" ::: "memory");   // wave-private LDS transpose fence
        FragU pf;
        const unsigned short* pp = lp + nn * 32 + hf * 8;
        pf.q[0] = *(const u32x4*)(pp);
        pf.q[1] = *(const u32x4*)(pp + 16);
#pragma unroll
        for (int t = 0; t < 4; t++) {
            bf16x16 vf = load_b_frag_g(Vh, S_LEN, t * 16, kbase);
            o[t] = wmma_bf16(pf.v, vf, o[t]);
        }
        asm volatile("s_wait_dscnt 0" ::: "memory");
    }
#pragma unroll
    for (int r = 0; r < 8; r++) {
        float invl = 1.0f / lrow[r];
        size_t tok = (size_t)(b * S_LEN + qbase + r + hf * 8);
#pragma unroll
        for (int t = 0; t < 4; t++)
            Ob[tok * QSIZE + h * HEAD_DIM + t * 16 + nn] = f2bf(o[t][r] * invl);
    }
}

extern "C" void kernel_launch(void* const* d_in, const int* in_sizes, int n_in,
                              void* d_out, int out_size, void* d_ws, size_t ws_size,
                              hipStream_t stream) {
    const int*   positions = (const int*)d_in[0];
    const float* hidden    = (const float*)d_in[1];
    const float* w_qkv     = (const float*)d_in[2];
    const float* w_o       = (const float*)d_in[3];
    const float* q_norm_w  = (const float*)d_in[4];
    const float* k_norm_w  = (const float*)d_in[5];
    float* out = (float*)d_out;

    char* ws = (char*)d_ws;
    unsigned short* hb    = (unsigned short*)(ws);               // 16 MB  hidden bf16
    unsigned short* wqkvb = (unsigned short*)(ws + 16777216);    // 12 MB  w_qkv bf16
    unsigned short* wob   = (unsigned short*)(ws + 29360128);    //  8 MB  w_o bf16
    float*          qkvf  = (float*)         (ws + 37748736);    // 48 MB  qkv f32
    unsigned short* Qb    = (unsigned short*)(ws + 88080384);    // 16 MB  Q bf16 [B,H,S,64]
    unsigned short* Kb    = (unsigned short*)(ws + 104857600);   //  4 MB  K bf16 [B,KV,S,64]
    unsigned short* Vt    = (unsigned short*)(ws + 109051904);   //  4 MB  V bf16 [B,KV,64,S]
    unsigned short* Ob    = (unsigned short*)(ws + 113246208);   // 16 MB  O bf16 [NTOK,QSIZE]

    k_f32_to_bf16<<<4096, 256, 0, stream>>>(hidden, hb, NTOK * HIDDEN);
    k_f32_to_bf16<<<4096, 256, 0, stream>>>(w_qkv, wqkvb, E_QKV * HIDDEN);
    k_f32_to_bf16<<<4096, 256, 0, stream>>>(w_o, wob, HIDDEN * QSIZE);

    // qkv[t,e] = sum_d hidden[t,d] * w_qkv[e,d]   (4096 x 3072 x 2048)
    // 64x48 = 3072 wave-tiles, 4 waves/block
    k_gemm_bf16<<<768, 128, 0, stream>>>(hb, wqkvb, qkvf, NTOK, E_QKV, HIDDEN);

    // RMSNorm + RoPE + scale fold + layout shuffle: 4096*48 wave-units, 8 waves/block
    k_norm_rope<<<24576, 256, 0, stream>>>(positions, qkvf, q_norm_w, k_norm_w, Qb, Kb, Vt);

    // causal GQA attention: 2*32*128 = 8192 waves, 4 waves/block
    k_attention<<<2048, 128, 0, stream>>>(Qb, Kb, Vt, Ob);

    // out[t,d] = sum_e O[t,e] * w_o[d,e]   (4096 x 2048 x 2048), f32 out
    // 64x32 = 2048 wave-tiles, 4 waves/block
    k_gemm_bf16<<<512, 128, 0, stream>>>(Ob, wob, out, NTOK, HIDDEN, QSIZE);
}
